// CausalSelfAttentionPyTorch_12249246728999
// MI455X (gfx1250) — matrix operand order repaired and checked by reading, not compiled, not run on hardware
//
#include <hip/hip_runtime.h>

// ---------------------------------------------------------------------------
// Causal self-attention (GPT-style) for MI455X / gfx1250.
// All GEMMs via v_wmma_f32_16x16x32_bf16 (bf16 inputs, f32 accumulate).
// Pipeline: [convert/transpose] -> [QKV GEMM] -> [flash attention] -> [proj].
// GEMMs use 32x64 per-wave tiles (8 WMMA / k-step) to halve L2 operand
// traffic vs 16x64 (A operands reused across 4 N-subtiles).
// ---------------------------------------------------------------------------

typedef __attribute__((ext_vector_type(16))) __bf16 v16bf;
typedef __attribute__((ext_vector_type(8)))  __bf16 v8bf;
typedef __attribute__((ext_vector_type(8)))  float  v8f;

#define D_MODEL   1024
#define N_HEAD    16
#define HEAD_DIM  64
#define SEQ       2048
#define BATCH     2
#define ROWS      (BATCH * SEQ)      // 4096
#define N_QKV     (3 * D_MODEL)      // 3072
#define NEG_INF   (-__builtin_inff())

// Workspace layout (bf16 element offsets)
static constexpr size_t OFF_XB  = 0;                                   // x  bf16 [4096,1024]
static constexpr size_t OFF_WAT = OFF_XB  + (size_t)ROWS   * D_MODEL;  // W_attn^T bf16 [3072,1024]
static constexpr size_t OFF_WPT = OFF_WAT + (size_t)N_QKV  * D_MODEL;  // W_proj^T bf16 [1024,1024]
static constexpr size_t OFF_QB  = OFF_WPT + (size_t)D_MODEL* D_MODEL;  // Q bf16 [B,H,T,Dh] (scaled)
static constexpr size_t OFF_KB  = OFF_QB  + (size_t)ROWS   * D_MODEL;  // K bf16 [B,H,T,Dh]
static constexpr size_t OFF_VT  = OFF_KB  + (size_t)ROWS   * D_MODEL;  // V^T bf16 [B,H,Dh,T]
static constexpr size_t OFF_YB  = OFF_VT  + (size_t)ROWS   * D_MODEL;  // y bf16 [4096,1024]
// total = 25,165,824 bf16 elements = 48 MB

// ---- WMMA operand loaders (ISA 7.12.2 layouts, wave32) --------------------
// A operand (16x32 bf16): lane holds row lane%16; K chunks at 16*half and 16*half+8 (contiguous).
__device__ __forceinline__ v16bf load_a2(const __bf16* p0, const __bf16* p1) {
  v8bf lo = *(const v8bf*)p0;
  v8bf hi = *(const v8bf*)p1;
  v16bf r;
#pragma unroll
  for (int i = 0; i < 8; ++i) { r[i] = lo[i]; r[i + 8] = hi[i]; }
  return r;
}
// B operand (32x16 bf16): lane holds column lane%16; K = 16*half + 0..15 contiguous.
__device__ __forceinline__ v16bf load_b32(const __bf16* p) {
  return *(const v16bf*)p;
}
__device__ __forceinline__ v8f wmma_bf16(v16bf a, v16bf b, v8f c) {
  return __builtin_amdgcn_wmma_f32_16x16x32_bf16(false, a, false, b, (short)0, c,
                                                 false, false);
}

// ---- Prep: fp32 -> bf16 conversions / weight transposes -------------------
__global__ __launch_bounds__(256) void k_prep_x(const float* __restrict__ src,
                                                __bf16* __restrict__ dst) {
  int i = blockIdx.x * 256 + threadIdx.x;          // 4,194,304 elements
  dst[i] = (__bf16)src[i];
}
__global__ __launch_bounds__(256) void k_prep_wa(const float* __restrict__ W,
                                                 __bf16* __restrict__ wt) {
  int i = blockIdx.x * 256 + threadIdx.x;          // 3,145,728 elements
  int n = i >> 10, k = i & 1023;                   // wt[n][k] = W[k][n]
  wt[i] = (__bf16)W[(size_t)k * N_QKV + n];
}
__global__ __launch_bounds__(256) void k_prep_wp(const float* __restrict__ W,
                                                 __bf16* __restrict__ wt) {
  int i = blockIdx.x * 256 + threadIdx.x;          // 1,048,576 elements
  int n = i >> 10, k = i & 1023;
  wt[i] = (__bf16)W[(size_t)k * D_MODEL + n];
}

// ---- GEMM1: qkv = x @ W_attn + b_attn ; scatter to Q (scaled), K, V^T -----
// 32x64 tile per wave: A operands (2 row-tiles) reused across 4 N-subtiles.
__global__ __launch_bounds__(256) void k_gemm_qkv(
    const __bf16* __restrict__ xb, const __bf16* __restrict__ wat,
    const float* __restrict__ b_attn, __bf16* __restrict__ Qb,
    __bf16* __restrict__ Kb, __bf16* __restrict__ Vt) {
  const int wid  = blockIdx.x * 8 + (threadIdx.x >> 5);
  const int lane = threadIdx.x & 31;
  const int half = lane >> 4, lm = lane & 15;
  const int mt = wid % (ROWS / 32);                // 0..127 (32 rows each)
  const int ng = wid / (ROWS / 32);                // 0..47  (64 cols each)
  if (ng >= N_QKV / 64) return;                    // uniform per wave

  v8f z = {};
  v8f acc[2][4] = {{z, z, z, z}, {z, z, z, z}};
  const __bf16* arow0 = xb + (size_t)(mt * 32 + lm) * D_MODEL;
  const __bf16* arow1 = arow0 + (size_t)16 * D_MODEL;
  const __bf16* brow[4];
#pragma unroll
  for (int t = 0; t < 4; ++t)
    brow[t] = wat + (size_t)(ng * 64 + t * 16 + lm) * D_MODEL;

  for (int k0 = 0; k0 < D_MODEL; k0 += 32) {
    v16bf a0 = load_a2(arow0 + k0 + 16 * half, arow0 + k0 + 16 * half + 8);
    v16bf a1 = load_a2(arow1 + k0 + 16 * half, arow1 + k0 + 16 * half + 8);
#pragma unroll
    for (int t = 0; t < 4; ++t) {
      v16bf bmat = load_b32(brow[t] + k0 + 16 * half);
      acc[0][t] = wmma_bf16(a0, bmat, acc[0][t]);
      acc[1][t] = wmma_bf16(a1, bmat, acc[1][t]);
    }
  }

#pragma unroll
  for (int t = 0; t < 4; ++t) {
    const int ncol = ng * 64 + t * 16 + lm;
    const float bias = b_attn[ncol];
#pragma unroll
    for (int r = 0; r < 2; ++r)
#pragma unroll
      for (int j = 0; j < 8; ++j) {
        const int m  = mt * 32 + r * 16 + j + 8 * half;  // global row [0,4096)
        const int bb = m >> 11, tt = m & (SEQ - 1);
        const float v = acc[r][t][j] + bias;
        if (ncol < D_MODEL) {                      // Q, fold softmax scale
          const int h = ncol >> 6, d = ncol & 63;
          Qb[(((size_t)(bb * N_HEAD + h)) * SEQ + tt) * HEAD_DIM + d] =
              (__bf16)(v * 0.125f);                // 1/sqrt(64)
        } else if (ncol < 2 * D_MODEL) {           // K
          const int nc = ncol - D_MODEL, h = nc >> 6, d = nc & 63;
          Kb[(((size_t)(bb * N_HEAD + h)) * SEQ + tt) * HEAD_DIM + d] =
              (__bf16)v;
        } else {                                   // V stored transposed
          const int nc = ncol - 2 * D_MODEL, h = nc >> 6, d = nc & 63;
          Vt[(((size_t)(bb * N_HEAD + h)) * HEAD_DIM + d) * SEQ + tt] =
              (__bf16)v;
        }
      }
  }
}

// ---- Flash attention: one wave per 16-query tile per (b,h) ----------------
__global__ __launch_bounds__(128) void k_attn(const __bf16* __restrict__ Qb,
                                              const __bf16* __restrict__ Kb,
                                              const __bf16* __restrict__ Vt,
                                              __bf16* __restrict__ yb) {
  __shared__ __align__(32) __bf16 pbuf[4][16 * 32];  // per-wave P staging
  const int w    = threadIdx.x >> 5;
  const int wid  = blockIdx.x * 4 + w;
  const int lane = threadIdx.x & 31;
  const int half = lane >> 4, lm = lane & 15;
  const int bh = wid >> 7;                           // (b*16+h)
  const int qt = wid & 127;
  const int q0 = qt * 16;
  const int b = bh >> 4, h = bh & 15;

  // Q A-operands (K-dim = Dh = 64 -> two 16x32 operands), loop-invariant
  const __bf16* qrow = Qb + ((size_t)bh * SEQ + q0 + lm) * HEAD_DIM;
  const v16bf aq0 = load_a2(qrow + 16 * half, qrow + 16 * half + 8);
  const v16bf aq1 = load_a2(qrow + 32 + 16 * half, qrow + 40 + 16 * half);

  v8f z = {};
  v8f acc[4] = {z, z, z, z};                         // 16x64 output in f32
  float mrow[8], lrow[8];
#pragma unroll
  for (int j = 0; j < 8; ++j) { mrow[j] = NEG_INF; lrow[j] = 0.f; }

  const int kbmax = (q0 + 15) >> 5;                  // 32-key blocks (causal)
  for (int kb = 0; kb <= kbmax; ++kb) {
    const int kc0 = kb * 32;
    // ---- S = Q K^T (two 16x16 tiles, K-dim = Dh via 2 WMMAs each) ----
    const __bf16* krow0 = Kb + ((size_t)bh * SEQ + kc0 + lm) * HEAD_DIM;
    const __bf16* krow1 = krow0 + 16 * HEAD_DIM;
    v8f s0 = z, s1 = z;
    s0 = wmma_bf16(aq0, load_b32(krow0 + 16 * half), s0);
    s0 = wmma_bf16(aq1, load_b32(krow0 + 32 + 16 * half), s0);
    s1 = wmma_bf16(aq0, load_b32(krow1 + 16 * half), s1);
    s1 = wmma_bf16(aq1, load_b32(krow1 + 32 + 16 * half), s1);

    // ---- online softmax update (row stats across 16-lane halves) ----
    const int kg0 = kc0 + lm, kg1 = kg0 + 16;
#pragma unroll
    for (int j = 0; j < 8; ++j) {
      const int qg = q0 + j + 8 * half;
      float a0 = (kg0 <= qg) ? s0[j] : NEG_INF;      // causal mask
      float a1 = (kg1 <= qg) ? s1[j] : NEG_INF;
      float r = fmaxf(a0, a1);
      r = fmaxf(r, __shfl_xor(r, 1, 16));
      r = fmaxf(r, __shfl_xor(r, 2, 16));
      r = fmaxf(r, __shfl_xor(r, 4, 16));
      r = fmaxf(r, __shfl_xor(r, 8, 16));
      const float mn = fmaxf(mrow[j], r);
      const float alpha = __expf(mrow[j] - mn);
      mrow[j] = mn;
      const float p0 = __expf(a0 - mn);
      const float p1 = __expf(a1 - mn);
      float rs = p0 + p1;
      rs += __shfl_xor(rs, 1, 16);
      rs += __shfl_xor(rs, 2, 16);
      rs += __shfl_xor(rs, 4, 16);
      rs += __shfl_xor(rs, 8, 16);
      lrow[j] = lrow[j] * alpha + rs;
#pragma unroll
      for (int t = 0; t < 4; ++t) acc[t][j] = acc[t][j] * alpha;
      // stage P (16x32 bf16, row-major) for the A-operand transpose
      pbuf[w][(j + 8 * half) * 32 + lm]      = (__bf16)p0;
      pbuf[w][(j + 8 * half) * 32 + 16 + lm] = (__bf16)p1;
    }

    // ---- O += P @ V  (A = 16x32 P from LDS, B = V^T rows, 4 Dh tiles) ----
    const v16bf ap =
        load_a2(&pbuf[w][lm * 32 + 16 * half], &pbuf[w][lm * 32 + 16 * half + 8]);
#pragma unroll
    for (int t = 0; t < 4; ++t) {
      const __bf16* vrow =
          Vt + ((size_t)(bh * HEAD_DIM + t * 16 + lm)) * SEQ + kc0 + 16 * half;
      acc[t] = wmma_bf16(ap, load_b32(vrow), acc[t]);
    }
  }

  // ---- normalize and store y (merged heads, row-major bf16) ----
  float inv[8];
#pragma unroll
  for (int j = 0; j < 8; ++j) inv[j] = 1.0f / lrow[j];
#pragma unroll
  for (int t = 0; t < 4; ++t)
#pragma unroll
    for (int j = 0; j < 8; ++j) {
      const size_t row = (size_t)b * SEQ + q0 + j + 8 * half;
      yb[row * D_MODEL + h * HEAD_DIM + t * 16 + lm] =
          (__bf16)(acc[t][j] * inv[j]);
    }
}

// ---- GEMM2: out = y @ W_proj + b_proj (f32 output), 32x64 tiles -----------
__global__ __launch_bounds__(256) void k_gemm_proj(
    const __bf16* __restrict__ ybm, const __bf16* __restrict__ wpt,
    const float* __restrict__ b_proj, float* __restrict__ out) {
  const int wid  = blockIdx.x * 8 + (threadIdx.x >> 5);
  const int lane = threadIdx.x & 31;
  const int half = lane >> 4, lm = lane & 15;
  const int mt = wid % (ROWS / 32);                // 0..127
  const int ng = wid / (ROWS / 32);                // 0..15
  if (ng >= D_MODEL / 64) return;

  v8f z = {};
  v8f acc[2][4] = {{z, z, z, z}, {z, z, z, z}};
  const __bf16* arow0 = ybm + (size_t)(mt * 32 + lm) * D_MODEL;
  const __bf16* arow1 = arow0 + (size_t)16 * D_MODEL;
  const __bf16* brow[4];
#pragma unroll
  for (int t = 0; t < 4; ++t)
    brow[t] = wpt + (size_t)(ng * 64 + t * 16 + lm) * D_MODEL;

  for (int k0 = 0; k0 < D_MODEL; k0 += 32) {
    v16bf a0 = load_a2(arow0 + k0 + 16 * half, arow0 + k0 + 16 * half + 8);
    v16bf a1 = load_a2(arow1 + k0 + 16 * half, arow1 + k0 + 16 * half + 8);
#pragma unroll
    for (int t = 0; t < 4; ++t) {
      v16bf bmat = load_b32(brow[t] + k0 + 16 * half);
      acc[0][t] = wmma_bf16(a0, bmat, acc[0][t]);
      acc[1][t] = wmma_bf16(a1, bmat, acc[1][t]);
    }
  }

#pragma unroll
  for (int t = 0; t < 4; ++t) {
    const int ncol = ng * 64 + t * 16 + lm;
    const float bias = b_proj[ncol];
#pragma unroll
    for (int r = 0; r < 2; ++r)
#pragma unroll
      for (int j = 0; j < 8; ++j) {
        const int m = mt * 32 + r * 16 + j + 8 * half;
        out[(size_t)m * D_MODEL + ncol] = acc[r][t][j] + bias;
      }
  }
}

// ---------------------------------------------------------------------------
extern "C" void kernel_launch(void* const* d_in, const int* in_sizes, int n_in,
                              void* d_out, int out_size, void* d_ws,
                              size_t ws_size, hipStream_t stream) {
  const float* x      = (const float*)d_in[0];
  const float* W_attn = (const float*)d_in[1];
  const float* b_attn = (const float*)d_in[2];
  const float* W_proj = (const float*)d_in[3];
  const float* b_proj = (const float*)d_in[4];
  float* out = (float*)d_out;

  __bf16* ws  = (__bf16*)d_ws;
  __bf16* xb  = ws + OFF_XB;
  __bf16* wat = ws + OFF_WAT;
  __bf16* wpt = ws + OFF_WPT;
  __bf16* Qb  = ws + OFF_QB;
  __bf16* Kb  = ws + OFF_KB;
  __bf16* Vt  = ws + OFF_VT;
  __bf16* yb  = ws + OFF_YB;

  k_prep_x <<<(ROWS * D_MODEL) / 256, 256, 0, stream>>>(x, xb);
  k_prep_wa<<<(N_QKV * D_MODEL) / 256, 256, 0, stream>>>(W_attn, wat);
  k_prep_wp<<<(D_MODEL * D_MODEL) / 256, 256, 0, stream>>>(W_proj, wpt);

  // 32x64 tiles: (4096/32)*(3072/64) = 6144 waves -> 768 blocks of 8 waves
  k_gemm_qkv<<<(ROWS / 32) * (N_QKV / 64) / 8, 256, 0, stream>>>(
      xb, wat, b_attn, Qb, Kb, Vt);

  k_attn<<<(BATCH * N_HEAD * (SEQ / 16)) / 4, 128, 0, stream>>>(Qb, Kb, Vt, yb);

  // (4096/32)*(1024/64) = 2048 waves -> 256 blocks of 8 waves
  k_gemm_proj<<<(ROWS / 32) * (D_MODEL / 64) / 8, 256, 0, stream>>>(
      yb, wpt, b_proj, out);
}
